// QKVAttentionWithRelativePosition_66967130079401
// MI455X (gfx1250) — matrix operand order repaired and checked by reading, not compiled, not run on hardware
//
#include <hip/hip_runtime.h>
#include <hip/hip_bf16.h>

typedef __attribute__((ext_vector_type(16))) __bf16 v16bf;
typedef __attribute__((ext_vector_type(8)))  float  v8f;

#define NHEADS 8
#define DHEAD  64
#define SEQ    512
#define QT     64     // query rows per workgroup
#define NW     4      // waves per workgroup (wave32)
#define MT     32     // key rows per iteration
#define EROWS  96     // union rel-band rows per iteration (MT + QT - 1 -> pad 96)
#define LDP    72     // LDS row stride (bf16) for [row][d] tiles

__global__ __launch_bounds__(128, 1)
void attn_rel_kernel(const float* __restrict__ qkv,
                     const float* __restrict__ kemb,
                     float* __restrict__ out)
{
    __shared__ __bf16 Qs[QT][LDP];         // Q tile, [l][d]
    __shared__ __bf16 Ks[MT][LDP];         // K tile, [m][d]
    __shared__ __bf16 Vs[MT][LDP];         // V tile, [m][d]
    __shared__ __bf16 Es[EROWS][LDP];      // rel-emb band, [c][d]
    __shared__ __bf16 Psc[NW][16][36];     // per-wave P re-layout scratch
    __shared__ float  Rsc[NW][16][48];     // per-wave rel-band scratch

    const int tid  = threadIdx.x;
    const int wave = tid >> 5;
    const int lane = tid & 31;
    const int n16  = lane & 15;            // N / column lane id
    const int g    = lane >> 4;            // half-wave group

    const int blk  = blockIdx.x;           // B*H*(SEQ/QT) = 512 blocks
    const int bh   = blk >> 3;
    const int qblk = blk & 7;
    const int b    = bh >> 3;
    const int h    = bh & 7;
    const int l0   = qblk * QT;
    const int lq   = l0 + wave * 16;

    const float* qkvB = qkv + (size_t)(b * NHEADS + h) * (3 * DHEAD) * SEQ; // [3D][L]
    const float* qP = qkvB;
    const float* kP = qkvB + (size_t)DHEAD * SEQ;
    const float* vP = qkvB + (size_t)(2 * DHEAD) * SEQ;
    const float* eP = kemb + (size_t)h * (2 * SEQ - 1) * DHEAD;             // [2L-1][D]

    // ---- load Q tile (fp32 global, d-major strided by L; coalesced over l) ----
    for (int i = 0; i < (QT * DHEAD) / 128; ++i) {
        int idx = i * 128 + tid;
        int d = idx >> 6;
        int l = idx & 63;
        Qs[l][d] = (__bf16)qP[(size_t)d * SEQ + (l0 + l)];
    }
    __syncthreads();

    // ---- build Q A-fragments (16x32 bf16): K = (e&7) + 8g + (e>=8 ? 16 : 0) ----
    v16bf Aq[2];
    {
        const int row = wave * 16 + n16;
        #pragma unroll
        for (int f = 0; f < 2; ++f)
            #pragma unroll
            for (int e = 0; e < 16; ++e) {
                int d = f * 32 + (e & 7) + 8 * g + ((e & 8) ? 16 : 0);
                Aq[f][e] = Qs[row][d];
            }
    }

    v8f O[4];                      // 16x64 output accumulator (4 d-tiles)
    #pragma unroll
    for (int f = 0; f < 4; ++f) O[f] = (v8f){0.f,0.f,0.f,0.f,0.f,0.f,0.f,0.f};
    float mrow[8], lrow[8];
    #pragma unroll
    for (int r = 0; r < 8; ++r) { mrow[r] = -1e30f; lrow[r] = 0.f; }

    for (int m0 = 0; m0 < SEQ; m0 += MT) {
        __syncthreads();           // previous iteration done with Ks/Vs/Es

        // ---- cooperative K / V tiles ----
        for (int i = 0; i < (MT * DHEAD) / 128; ++i) {
            int idx = i * 128 + tid;
            int m = idx & 31;
            int d = idx >> 5;
            Ks[m][d] = (__bf16)kP[(size_t)d * SEQ + (m0 + m)];
            Vs[m][d] = (__bf16)vP[(size_t)d * SEQ + (m0 + m)];
        }
        // ---- cooperative rel-emb band: rows [m0-l0+448 , +95] (clamped) ----
        const int ubase = m0 - l0 + (SEQ - QT);
        for (int i = 0; i < (EROWS * DHEAD) / 128; ++i) {
            int idx = i * 128 + tid;
            int d = idx & 63;
            int r = idx >> 6;
            int c = ubase + r;
            c = (c > 2 * SEQ - 2) ? (2 * SEQ - 2) : c;   // only never-gathered pad rows clamp
            Es[r][d] = (__bf16)eP[(size_t)c * DHEAD + d];
        }
        __syncthreads();

        // ---- S = Q * K^T  (two 16x16 tiles over m, K-dim = 64 = 2 chained WMMAs) ----
        v8f S0 = (v8f){0.f,0.f,0.f,0.f,0.f,0.f,0.f,0.f};
        v8f S1 = S0;
        #pragma unroll
        for (int kc = 0; kc < 2; ++kc) {
            v16bf B0, B1;
            #pragma unroll
            for (int e = 0; e < 16; ++e) {
                int d = kc * 32 + (e & 7) + 8 * g + ((e & 8) ? 16 : 0);            // B: K = (e&7) + 8g + (e>=8 ? 16 : 0)
                B0[e] = Ks[n16][d];
                B1[e] = Ks[16 + n16][d];
            }
            S0 = __builtin_amdgcn_wmma_f32_16x16x32_bf16(false, Aq[kc], false, B0, (short)0, S0, false, false);
            S1 = __builtin_amdgcn_wmma_f32_16x16x32_bf16(false, Aq[kc], false, B1, (short)0, S1, false, false);
        }

        // ---- rel band GEMM: R[16 x 48] = Q * E_band^T ----
        {
            const int eoff = (NW - 1 - wave) * 16;       // this wave's offset in union band
            v8f R0 = (v8f){0.f,0.f,0.f,0.f,0.f,0.f,0.f,0.f};
            v8f R1 = R0, R2 = R0;
            #pragma unroll
            for (int kc = 0; kc < 2; ++kc) {
                v16bf Be0, Be1, Be2;
                #pragma unroll
                for (int e = 0; e < 16; ++e) {
                    int d = kc * 32 + (e & 7) + 8 * g + ((e & 8) ? 16 : 0);
                    Be0[e] = Es[eoff +      n16][d];
                    Be1[e] = Es[eoff + 16 + n16][d];
                    Be2[e] = Es[eoff + 32 + n16][d];
                }
                R0 = __builtin_amdgcn_wmma_f32_16x16x32_bf16(false, Aq[kc], false, Be0, (short)0, R0, false, false);
                R1 = __builtin_amdgcn_wmma_f32_16x16x32_bf16(false, Aq[kc], false, Be1, (short)0, R1, false, false);
                R2 = __builtin_amdgcn_wmma_f32_16x16x32_bf16(false, Aq[kc], false, Be2, (short)0, R2, false, false);
            }
            #pragma unroll
            for (int r = 0; r < 8; ++r) {
                Rsc[wave][r + 8 * g][n16]      = R0[r];
                Rsc[wave][r + 8 * g][16 + n16] = R1[r];
                Rsc[wave][r + 8 * g][32 + n16] = R2[r];
            }
        }
        __syncthreads();

        // ---- shift-gather rel band: S[l][m] += R[l][ (m-m0) - (l-lq) + 15 ] ----
        #pragma unroll
        for (int r = 0; r < 8; ++r) {
            int rl = r + 8 * g;
            S0[r] += Rsc[wave][rl][n16 - rl + 15];       // idx in [0,30]
            S1[r] += Rsc[wave][rl][n16 + 31 - rl];       // idx in [16,46]
        }

        // ---- online softmax (row reductions across 16 lanes) ----
        float sc[8];
        #pragma unroll
        for (int r = 0; r < 8; ++r) {
            float v = fmaxf(S0[r], S1[r]);
            #pragma unroll
            for (int off = 1; off < 16; off <<= 1)
                v = fmaxf(v, __shfl_xor(v, off, 32));
            float mn = fmaxf(mrow[r], v);
            sc[r]   = __expf(mrow[r] - mn);
            mrow[r] = mn;
            S0[r] = __expf(S0[r] - mn);
            S1[r] = __expf(S1[r] - mn);
            float s = S0[r] + S1[r];
            #pragma unroll
            for (int off = 1; off < 16; off <<= 1)
                s += __shfl_xor(s, off, 32);
            lrow[r] = lrow[r] * sc[r] + s;
        }
        #pragma unroll
        for (int f = 0; f < 4; ++f)
            #pragma unroll
            for (int r = 0; r < 8; ++r)
                O[f][r] *= sc[r];

        // ---- re-layout P (D-frag f32 -> A-frag bf16) through LDS ----
        #pragma unroll
        for (int r = 0; r < 8; ++r) {
            Psc[wave][r + 8 * g][n16]      = (__bf16)S0[r];
            Psc[wave][r + 8 * g][16 + n16] = (__bf16)S1[r];
        }
        __syncthreads();
        v16bf Ap;
        #pragma unroll
        for (int e = 0; e < 16; ++e) {
            int mc = (e & 7) + 8 * g + ((e & 8) ? 16 : 0);
            Ap[e] = Psc[wave][n16][mc];
        }

        // ---- O += P * V  (K = 32 = full m-tile) ----
        #pragma unroll
        for (int f = 0; f < 4; ++f) {
            v16bf Bv;
            #pragma unroll
            for (int e = 0; e < 16; ++e)
                Bv[e] = Vs[(e & 7) + 8 * g + ((e & 8) ? 16 : 0)][f * 16 + n16];
            O[f] = __builtin_amdgcn_wmma_f32_16x16x32_bf16(false, Ap, false, Bv, (short)0, O[f], false, false);
        }
    }

    // ---- normalize and store: out is [B][H][L][D] contiguous (raw reshape) ----
    float* outP = out + ((size_t)(b * NHEADS + h) * SEQ + lq) * DHEAD;
    #pragma unroll
    for (int r = 0; r < 8; ++r) {
        float inv = 1.0f / lrow[r];
        int rl = r + 8 * g;
        #pragma unroll
        for (int f = 0; f < 4; ++f)
            outP[(size_t)rl * DHEAD + f * 16 + n16] = O[f][r] * inv;
    }
}

extern "C" void kernel_launch(void* const* d_in, const int* in_sizes, int n_in,
                              void* d_out, int out_size, void* d_ws, size_t ws_size,
                              hipStream_t stream) {
    const float* qkv  = (const float*)d_in[0];
    const float* kemb = (const float*)d_in[1];
    // d_in[2] (v_emb) is unused by the reference computation (add_relative_to_values=False)
    float* out = (float*)d_out;

    dim3 grid(8 * NHEADS * (SEQ / QT));   // 512 workgroups
    dim3 block(128);                      // 4 wave32 waves
    attn_rel_kernel<<<grid, block, 0, stream>>>(qkv, kemb, out);
}
